// SelfAttention_17377437680002
// MI455X (gfx1250) — matrix operand order repaired and checked by reading, not compiled, not run on hardware
//
#include <hip/hip_runtime.h>
#include <hip/hip_bf16.h>

// ---------------------------------------------------------------------------
// Self-attention with aperture banding for MI455X (gfx1250, wave32, WMMA).
//   1. cvt: f32 -> bf16 for x, Wq, Wk, Wv, Wo (workspace)
//   2. gemm: Qb = 0.06*(xb Wq^T), Kb = xb Wk^T, Vb = xb Wv^T   (bf16 out)
//      - double-buffered LDS tiles, async global->LDS (ASYNCcnt) staging
//   3. fill: zero the full [N,N] f32 attention output
//   4. attn: banded logits (WMMA) + softmax, writes att band (f32)
//   5. ypre: y_pre = att^T @ V over the band (WMMA, LDS-transposed operands)
//   6. gemm: y = y_pre Wo^T (f32 out)
// ---------------------------------------------------------------------------

typedef __attribute__((ext_vector_type(16))) __bf16 v16bf;
typedef __attribute__((ext_vector_type(8)))  __bf16 v8bf;
typedef __attribute__((ext_vector_type(8)))  float  v8f;
typedef int v4i __attribute__((vector_size(16)));

#define N_TOK  8192
#define DMODEL 1024
#define AP     250
#define LW     544   // 34 tiles of 16: covers worst-case band window (<=536)

#if __has_builtin(__builtin_amdgcn_global_load_async_to_lds_b128)
#define HAS_ASYNC_LDS 1
#else
#define HAS_ASYNC_LDS 0
#endif

// 16B global -> LDS copy; async (no VGPR round-trip, ASYNCcnt) when available.
static __device__ __forceinline__ void copy16_g2l(const void* g, void* l) {
#if HAS_ASYNC_LDS
  __builtin_amdgcn_global_load_async_to_lds_b128(
      (__attribute__((address_space(1))) v4i*)g,
      (__attribute__((address_space(3))) v4i*)l, 0, 0);
#else
  *(v8bf*)l = *(const v8bf*)g;
#endif
}

static __device__ __forceinline__ void wait_async() {
#if HAS_ASYNC_LDS
#if __has_builtin(__builtin_amdgcn_s_wait_asynccnt)
  __builtin_amdgcn_s_wait_asynccnt(0);
#else
  asm volatile("s_wait_asynccnt 0x0" ::: "memory");
#endif
#endif
}

static __device__ __forceinline__ v16bf cat8(v8bf lo, v8bf hi) {
  v16bf r;
#pragma unroll
  for (int e = 0; e < 8; ++e) { r[e] = lo[e]; r[8 + e] = hi[e]; }
  return r;
}

static __device__ __forceinline__ v8f wmma_bf16(v16bf a, v16bf b, v8f c) {
  // D = A(16x32 bf16) x B(32x16 bf16) + C(16x16 f32)
  return __builtin_amdgcn_wmma_f32_16x16x32_bf16(false, a, false, b,
                                                 (short)0, c, false, false);
}

// ---------------------------------------------------------------------------
__global__ void k_cvt_bf16(const float* __restrict__ in,
                           __bf16* __restrict__ out, int n) {
  int i = blockIdx.x * blockDim.x + threadIdx.x;
  int stride = gridDim.x * blockDim.x;
  for (; i < n; i += stride) out[i] = (__bf16)in[i];
}

__global__ void k_fill0(float* __restrict__ p, int n) {
  int i = blockIdx.x * blockDim.x + threadIdx.x;
  int stride = gridDim.x * blockDim.x;
  for (; i < n; i += stride) p[i] = 0.0f;
}

// ---------------------------------------------------------------------------
// C[M,Nout] = scale * A[M,K](bf16) @ W[Nout,K]^T(bf16), f32 accumulate.
// Block tile 64x128, 8 waves (2x4), 32x32 per wave (2x2 WMMA tiles).
// Double-buffered LDS tiles, row stride 40 elems (80B -> 16B-aligned chunks).
__global__ __launch_bounds__(256) void k_gemm_bf16(
    const __bf16* __restrict__ A, const __bf16* __restrict__ W,
    __bf16* __restrict__ outb, float* __restrict__ outf,
    float scale, int K, int Nout) {
  __shared__ __bf16 At[2][64 * 40];
  __shared__ __bf16 Bt[2][128 * 40];
  const int tid  = threadIdx.x;
  const int lane = tid & 31;
  const int w    = tid >> 5;
  const int wm   = w & 1;       // 0..1  (M dir)
  const int wn   = w >> 1;      // 0..3  (N dir)
  const int m0   = blockIdx.x * 64;
  const int n0   = blockIdx.y * 128;
  const int r16  = lane & 15;
  const int hsel = (lane & 16) ? 1 : 0;

  // staging geometry: A = 64 rows x 4 chunks (1/thread); B = 128 x 4 (2/thread)
  const int rowA = tid >> 2, segA = tid & 3;
  const int rowB = tid >> 1, segB = (tid & 1) * 2;

  v8f acc[2][2] = {};

  auto stage = [&](int buf, int k0) {
    copy16_g2l(A + (size_t)(m0 + rowA) * K + k0 + segA * 8,
               &At[buf][rowA * 40 + segA * 8]);
    copy16_g2l(W + (size_t)(n0 + rowB) * K + k0 + segB * 8,
               &Bt[buf][rowB * 40 + segB * 8]);
    copy16_g2l(W + (size_t)(n0 + rowB) * K + k0 + (segB + 1) * 8,
               &Bt[buf][rowB * 40 + (segB + 1) * 8]);
  };

  stage(0, 0);
  wait_async();
  __syncthreads();

  const int KS = K / 32;
  for (int ks = 0; ks < KS; ++ks) {
    const int cur = ks & 1;
    if (ks + 1 < KS) stage(cur ^ 1, (ks + 1) * 32);  // overlap with WMMA

#pragma unroll
    for (int mt = 0; mt < 2; ++mt) {
      // A fragment (lane = M; lanes<16 K0..15, lanes>=16 K16..31)
      const __bf16* ap = &At[cur][(wm * 32 + mt * 16 + r16) * 40 + hsel * 16];
      v16bf a = cat8(*(const v8bf*)ap, *(const v8bf*)(ap + 8));
#pragma unroll
      for (int nt = 0; nt < 2; ++nt) {
        // B fragment (lane = N; lanes<16 K0..15, lanes>=16 K16..31)
        const __bf16* bp = &Bt[cur][(wn * 32 + nt * 16 + r16) * 40 + hsel * 16];
        v16bf b = cat8(*(const v8bf*)bp, *(const v8bf*)(bp + 8));
        acc[mt][nt] = wmma_bf16(a, b, acc[mt][nt]);
      }
    }
    wait_async();
    __syncthreads();
  }

  const int mo = hsel * 8;
#pragma unroll
  for (int mt = 0; mt < 2; ++mt) {
#pragma unroll
    for (int nt = 0; nt < 2; ++nt) {
      const int col = n0 + wn * 32 + nt * 16 + r16;
#pragma unroll
      for (int v = 0; v < 8; ++v) {
        const int row = m0 + wm * 32 + mt * 16 + mo + v;
        const float val = acc[mt][nt][v] * scale;
        if (outb) outb[(size_t)row * Nout + col] = (__bf16)val;
        if (outf) outf[(size_t)row * Nout + col] = val;
      }
    }
  }
}

// ---------------------------------------------------------------------------
// Banded logits + softmax. One WG = 16 query rows; band window <= 34 N-tiles,
// round-robined over 8 waves (<=5 tiles each). Logits via WMMA, softmax in LDS.
__global__ __launch_bounds__(256) void k_attn(
    const __bf16* __restrict__ Qb, const __bf16* __restrict__ Kb,
    float* __restrict__ att) {
  __shared__ float lbuf[16 * LW];
  __shared__ float red[16 * 16];
  __shared__ float rowmax[16];
  __shared__ float rowsum[16];

  const int tid  = threadIdx.x;
  const int lane = tid & 31;
  const int w    = tid >> 5;
  const int i0   = blockIdx.x * 16;

  int lo = i0 - (AP - 1); if (lo < 0) lo = 0;
  int hi = i0 + 15 + AP;  if (hi > N_TOK) hi = N_TOK;
  const int jt0 = lo & ~15;
  const int NT  = (hi - jt0 + 15) >> 4;   // <= 34

  for (int idx = tid; idx < 16 * LW; idx += 256) lbuf[idx] = -__builtin_inff();
  __syncthreads();

  const int r16  = lane & 15;
  const int hsel = (lane & 16) ? 1 : 0;

  v8f acc[5] = {};
  for (int k0 = 0; k0 < DMODEL; k0 += 32) {
    const __bf16* ap = Qb + (size_t)(i0 + r16) * DMODEL + k0 + hsel * 16;
    v16bf a = cat8(*(const v8bf*)ap, *(const v8bf*)(ap + 8));
#pragma unroll
    for (int ti = 0; ti < 5; ++ti) {
      const int t = w + ti * 8;
      if (t < NT) {
        const int j = jt0 + t * 16 + r16;
        v16bf b = *(const v16bf*)(Kb + (size_t)j * DMODEL + k0 + hsel * 16);
        acc[ti] = wmma_bf16(a, b, acc[ti]);
      }
    }
  }

  // dump logits (with aperture mask) into LDS
#pragma unroll
  for (int ti = 0; ti < 5; ++ti) {
    const int t = w + ti * 8;
    if (t < NT) {
      const int col = t * 16 + r16;
      const int j = jt0 + col;
#pragma unroll
      for (int v = 0; v < 8; ++v) {
        const int m = hsel * 8 + v;
        int d = (i0 + m) - j; if (d < 0) d = -d;
        lbuf[m * LW + col] = (d < AP) ? acc[ti][v] : -__builtin_inff();
      }
    }
  }
  __syncthreads();

  // softmax: 16 threads per row
  const int r = tid >> 4, c = tid & 15;
  float mx = -__builtin_inff();
  for (int cc = c; cc < LW; cc += 16) mx = fmaxf(mx, lbuf[r * LW + cc]);
  red[r * 16 + c] = mx;
  __syncthreads();
  if (c == 0) {
    float m2 = -__builtin_inff();
    for (int q = 0; q < 16; ++q) m2 = fmaxf(m2, red[r * 16 + q]);
    rowmax[r] = m2;
  }
  __syncthreads();
  const float rm = rowmax[r];
  float s = 0.0f;
  for (int cc = c; cc < LW; cc += 16) s += __expf(lbuf[r * LW + cc] - rm);
  red[r * 16 + c] = s;
  __syncthreads();
  if (c == 0) {
    float s2 = 0.0f;
    for (int q = 0; q < 16; ++q) s2 += red[r * 16 + q];
    rowsum[r] = s2;
  }
  __syncthreads();
  const float inv = 1.0f / rowsum[r];
  for (int cc = c; cc < NT * 16; cc += 16) {
    att[(size_t)(i0 + r) * N_TOK + jt0 + cc] =
        __expf(lbuf[r * LW + cc] - rm) * inv;
  }
}

// ---------------------------------------------------------------------------
// y_pre[j,d] = sum_i att[i,j] * V[i,d]  (band: |i-j| < AP).
// One WG = 16 j-rows x 256 d-cols; grid (512, 4). Per k-step the WG stages:
//   Abuf[16][32]  = att^T tile (f32 gather coalesced along j, cvt bf16)
//   Vt[256][32]   = V^T tile   (v8bf row loads, 2B scatter into LDS)
// then 8 waves x 2 N-tiles consume them as WMMA fragments via ds_load_b128.
__global__ __launch_bounds__(256) void k_ypre(
    const float* __restrict__ att, const __bf16* __restrict__ Vb,
    __bf16* __restrict__ ypb) {
  __shared__ __bf16 Abuf[16 * 40];
  __shared__ __bf16 Vt[256 * 40];
  const int tid  = threadIdx.x;
  const int lane = tid & 31;
  const int w    = tid >> 5;
  const int jt   = blockIdx.x * 16;
  const int d0   = blockIdx.y * 256;

  int lo = jt - (AP - 1); if (lo < 0) lo = 0;
  int hi = jt + 15 + AP;  if (hi > N_TOK) hi = N_TOK;
  const int it0 = lo & ~15;
  const int KT  = (hi - it0 + 15) >> 4;  // <= 34 k-tiles of 16
  const int KS  = (KT + 1) >> 1;         // k-steps of 32

  const int r16  = lane & 15;
  const int hsel = (lane & 16) ? 1 : 0;

  v8f acc[2] = {};
  for (int ks = 0; ks < KS; ++ks) {
    const int ib = it0 + ks * 32;
    __syncthreads();  // previous fragments consumed before restaging

    // att^T tile: idx -> (j-local = idx&15, k = idx>>4); coalesced in j
#pragma unroll
    for (int t = 0; t < 2; ++t) {
      const int idx = tid + t * 256;
      const int m = idx & 15, k = idx >> 4;
      const int i = ib + k;
      const float av = (i < N_TOK) ? att[(size_t)i * N_TOK + jt + m] : 0.0f;
      Abuf[m * 40 + k] = (__bf16)av;
    }
    // V^T tile: idx -> (d8 = (idx&31)*8, i = idx>>5); v8bf row load, scatter
#pragma unroll
    for (int t = 0; t < 4; ++t) {
      const int idx = tid + t * 256;
      const int d8 = (idx & 31) * 8, i = idx >> 5;
      const int gi = ib + i;
      v8bf vv;
      if (gi < N_TOK) {
        vv = *(const v8bf*)(Vb + (size_t)gi * DMODEL + d0 + d8);
      } else {
#pragma unroll
        for (int e = 0; e < 8; ++e) vv[e] = (__bf16)0.0f;
      }
#pragma unroll
      for (int e = 0; e < 8; ++e) Vt[(d8 + e) * 40 + i] = vv[e];
    }
    __syncthreads();

    const __bf16* ap = Abuf + r16 * 40 + hsel * 16;
    v16bf a = cat8(*(const v8bf*)ap, *(const v8bf*)(ap + 8));
#pragma unroll
    for (int q = 0; q < 2; ++q) {
      const int dl = (w * 2 + q) * 16 + r16;
      const __bf16* bp = Vt + dl * 40 + hsel * 16;
      v16bf b = cat8(*(const v8bf*)bp, *(const v8bf*)(bp + 8));
      acc[q] = wmma_bf16(a, b, acc[q]);
    }
  }

#pragma unroll
  for (int q = 0; q < 2; ++q) {
    const int d = d0 + (w * 2 + q) * 16 + r16;
#pragma unroll
    for (int v = 0; v < 8; ++v) {
      const int row = jt + hsel * 8 + v;
      ypb[(size_t)row * DMODEL + d] = (__bf16)acc[q][v];
    }
  }
}

// ---------------------------------------------------------------------------
extern "C" void kernel_launch(void* const* d_in, const int* in_sizes, int n_in,
                              void* d_out, int out_size, void* d_ws,
                              size_t ws_size, hipStream_t stream) {
  const float* x  = (const float*)d_in[0];
  const float* Wq = (const float*)d_in[1];
  const float* Wk = (const float*)d_in[2];
  const float* Wv = (const float*)d_in[3];
  const float* Wo = (const float*)d_in[4];

  float* y_out = (float*)d_out;                        // [N, D] f32
  float* att   = y_out + (size_t)N_TOK * DMODEL;       // [N, N] f32

  char* ws = (char*)d_ws;
  size_t off = 0;
  auto walloc = [&](size_t bytes) -> char* {
    char* p = ws + off;
    off += (bytes + 255) & ~(size_t)255;
    return p;
  };
  __bf16* xb  = (__bf16*)walloc((size_t)N_TOK * DMODEL * 2);
  __bf16* Wqb = (__bf16*)walloc((size_t)DMODEL * DMODEL * 2);
  __bf16* Wkb = (__bf16*)walloc((size_t)DMODEL * DMODEL * 2);
  __bf16* Wvb = (__bf16*)walloc((size_t)DMODEL * DMODEL * 2);
  __bf16* Wob = (__bf16*)walloc((size_t)DMODEL * DMODEL * 2);
  __bf16* Qb  = (__bf16*)walloc((size_t)N_TOK * DMODEL * 2);
  __bf16* Kb  = (__bf16*)walloc((size_t)N_TOK * DMODEL * 2);
  __bf16* Vb  = (__bf16*)walloc((size_t)N_TOK * DMODEL * 2);
  __bf16* ypb = (__bf16*)walloc((size_t)N_TOK * DMODEL * 2);

  // 1. convert to bf16
  k_cvt_bf16<<<512, 256, 0, stream>>>(x, xb, N_TOK * DMODEL);
  k_cvt_bf16<<<256, 256, 0, stream>>>(Wq, Wqb, DMODEL * DMODEL);
  k_cvt_bf16<<<256, 256, 0, stream>>>(Wk, Wkb, DMODEL * DMODEL);
  k_cvt_bf16<<<256, 256, 0, stream>>>(Wv, Wvb, DMODEL * DMODEL);
  k_cvt_bf16<<<256, 256, 0, stream>>>(Wo, Wob, DMODEL * DMODEL);

  // 2. Q/K/V projections (Q pre-scaled by 0.06 as in reference)
  dim3 gg(N_TOK / 64, DMODEL / 128);
  k_gemm_bf16<<<gg, 256, 0, stream>>>(xb, Wqb, Qb, nullptr, 0.06f, DMODEL, DMODEL);
  k_gemm_bf16<<<gg, 256, 0, stream>>>(xb, Wkb, Kb, nullptr, 1.0f, DMODEL, DMODEL);
  k_gemm_bf16<<<gg, 256, 0, stream>>>(xb, Wvb, Vb, nullptr, 1.0f, DMODEL, DMODEL);

  // 3. zero the full att output (band kernel writes only its window)
  k_fill0<<<4096, 256, 0, stream>>>(att, N_TOK * N_TOK);

  // 4. banded logits + softmax
  k_attn<<<N_TOK / 16, 256, 0, stream>>>(Qb, Kb, att);

  // 5. y_pre = att^T @ V over the band
  dim3 gy(N_TOK / 16, DMODEL / 256);
  k_ypre<<<gy, 256, 0, stream>>>(att, Vb, ypb);

  // 6. y = y_pre @ Wo^T (f32 out)
  k_gemm_bf16<<<gg, 256, 0, stream>>>(ypb, Wob, nullptr, y_out, 1.0f, DMODEL, DMODEL);
}
